// FeatureExtractor_49572512530931
// MI455X (gfx1250) — compile-verified, matrix operand order checked by reading, not run on hardware
//
#include <hip/hip_runtime.h>
#include <hip/hip_bf16.h>
#include <math.h>

typedef __attribute__((ext_vector_type(16))) _Float16 v16h;
typedef __attribute__((ext_vector_type(8)))  _Float16 v8h;
typedef __attribute__((ext_vector_type(8)))  float    v8f;
typedef __attribute__((ext_vector_type(4)))  float    v4f;

static constexpr int BATCH = 2;
static constexpr int NPTS  = 8192;
static constexpr int NP1   = 1024;
static constexpr int NP2   = 256;
static constexpr int KNB   = 16;   // knn for SA / transformer
static constexpr int NORMK = 10;   // knn for normal estimation

static inline int cdiv(int a, int b) { return (a + b - 1) / b; }

// ---------------------------------------------------------------------------
// small device helpers
// ---------------------------------------------------------------------------
__device__ __forceinline__ void cross3(const float* a, const float* b, float* c) {
  c[0] = a[1] * b[2] - a[2] * b[1];
  c[1] = a[2] * b[0] - a[0] * b[2];
  c[2] = a[0] * b[1] - a[1] * b[0];
}
__device__ __forceinline__ float dot3(const float* a, const float* b) {
  return a[0] * b[0] + a[1] * b[1] + a[2] * b[2];
}

// smallest eigenvector of symmetric 3x3 (analytic trig method)
__device__ void smallest_eigvec3(float xx, float xy, float xz, float yy, float yz,
                                 float zz, float* v) {
  float p1 = xy * xy + xz * xz + yz * yz;
  if (p1 < 1e-18f) {
    v[0] = v[1] = v[2] = 0.f;
    if (xx <= yy && xx <= zz) v[0] = 1.f;
    else if (yy <= zz) v[1] = 1.f;
    else v[2] = 1.f;
    return;
  }
  float q = (xx + yy + zz) * (1.f / 3.f);
  float a = xx - q, b = yy - q, c = zz - q;
  float p2 = a * a + b * b + c * c + 2.f * p1;
  float p = sqrtf(p2 * (1.f / 6.f));
  float inv = 1.f / p;
  float bxx = a * inv, bxy = xy * inv, bxz = xz * inv;
  float byy = b * inv, byz = yz * inv, bzz = c * inv;
  float detB = bxx * (byy * bzz - byz * byz) - bxy * (bxy * bzz - byz * bxz) +
               bxz * (bxy * byz - byy * bxz);
  float r = fminf(1.f, fmaxf(-1.f, 0.5f * detB));
  float phi = acosf(r) * (1.f / 3.f);
  float lam = q + 2.f * p * cosf(phi + 2.0943951023931953f);  // smallest eigval
  float r0[3] = {xx - lam, xy, xz};
  float r1[3] = {xy, yy - lam, yz};
  float r2[3] = {xz, yz, zz - lam};
  float c01[3], c02[3], c12[3];
  cross3(r0, r1, c01); cross3(r0, r2, c02); cross3(r1, r2, c12);
  float d01 = dot3(c01, c01), d02 = dot3(c02, c02), d12 = dot3(c12, c12);
  const float* best = c01; float db = d01;
  if (d02 > db) { best = c02; db = d02; }
  if (d12 > db) { best = c12; db = d12; }
  float s = rsqrtf(fmaxf(db, 1e-30f));
  v[0] = best[0] * s; v[1] = best[1] * s; v[2] = best[2] * s;
}

// ---------------------------------------------------------------------------
// KNN: LDS-tiled reference set (each 128-point tile is reused by all 128
// queries of the block). Top-K kept in VGPRs: insert at the tail slot and
// bubble up with a constant-index compare-swap ladder (no scratch).
// Blocks never straddle a batch (all Mq are multiples of 128).
// ---------------------------------------------------------------------------
template <int K>
__global__ void knn_kernel(const float* __restrict__ qry, const float* __restrict__ ref,
                           int Mq, int Nr, int C, int* __restrict__ oidx,
                           float* __restrict__ odist) {
  constexpr int TILE = 128;
  __shared__ float tile[TILE * 6];
  int g = blockIdx.x * blockDim.x + threadIdx.x;
  bool valid = g < BATCH * Mq;
  int gq = valid ? g : (BATCH * Mq - 1);
  int b = gq / Mq;
  const float* qp = qry + (size_t)gq * C;
  float qv[6];
#pragma unroll
  for (int c = 0; c < 6; ++c) {
    int cc = (c < C) ? c : 0;      // clamped -> always dereferenceable
    float t = qp[cc];              // unconditional load
    qv[c] = (c < C) ? t : 0.f;     // branchless select
  }
  float bd[K]; int bi[K];
#pragma unroll
  for (int i = 0; i < K; ++i) { bd[i] = 1e30f; bi[i] = 0; }
  const float* rb = ref + (size_t)b * Nr * C;
  for (int t0 = 0; t0 < Nr; t0 += TILE) {
    int cnt = min(TILE, Nr - t0);
    __syncthreads();
    for (int i = threadIdx.x; i < cnt * C; i += (int)blockDim.x)
      tile[i] = rb[(size_t)t0 * C + i];
    __syncthreads();
    for (int n = 0; n < cnt; ++n) {
      float d = 0.f;
      for (int c = 0; c < C; ++c) { float t = qv[c] - tile[n * C + c]; d = fmaf(t, t, d); }
      if (d < bd[K - 1]) {
        bd[K - 1] = d; bi[K - 1] = t0 + n;
#pragma unroll
        for (int i = K - 1; i > 0; --i) {
          bool sw = bd[i] < bd[i - 1];
          float td = bd[i]; int ti = bi[i];
          bd[i]     = sw ? bd[i - 1] : bd[i];
          bi[i]     = sw ? bi[i - 1] : bi[i];
          bd[i - 1] = sw ? td : bd[i - 1];
          bi[i - 1] = sw ? ti : bi[i - 1];
        }
      }
    }
  }
  if (valid) {
    int* op = oidx + (size_t)g * K;
#pragma unroll
    for (int i = 0; i < K; ++i) op[i] = bi[i];
    if (odist) {
      float* dp = odist + (size_t)g * K;
#pragma unroll
      for (int i = 0; i < K; ++i) dp[i] = bd[i];
    }
  }
}

// ---------------------------------------------------------------------------
// Normal estimation (tangent frames X/Y are dead code w.r.t. outputs -> elided)
// ---------------------------------------------------------------------------
__global__ void cal_axis_kernel(const float* __restrict__ xyz, const int* __restrict__ nidx,
                                float* __restrict__ nrm) {
  int g = blockIdx.x * blockDim.x + threadIdx.x;
  if (g >= BATCH * NPTS) return;
  int b = g / NPTS;
  const float* xb = xyz + (size_t)b * NPTS * 3;
  const float* p = xyz + (size_t)g * 3;
  const int* nn = nidx + (size_t)g * NORMK;
  float gx[NORMK], gy[NORMK], gz[NORMK];
  float mx = 0.f, my = 0.f, mz = 0.f;
#pragma unroll
  for (int i = 0; i < NORMK; ++i) {
    const float* t = xb + (size_t)nn[i] * 3;
    gx[i] = t[0] - p[0]; gy[i] = t[1] - p[1]; gz[i] = t[2] - p[2];
    mx += gx[i]; my += gy[i]; mz += gz[i];
  }
  const float inv = 1.f / (float)NORMK;
  mx *= inv; my *= inv; mz *= inv;
  float xx = 0.f, xy = 0.f, xz = 0.f, yy = 0.f, yz = 0.f, zz = 0.f;
#pragma unroll
  for (int i = 0; i < NORMK; ++i) {
    float a = gx[i] - mx, b2 = gy[i] - my, c2 = gz[i] - mz;
    xx += a * a; xy += a * b2; xz += a * c2;
    yy += b2 * b2; yz += b2 * c2; zz += c2 * c2;
  }
  xx *= inv; xy *= inv; xz *= inv; yy *= inv; yz *= inv; zz *= inv;
  float v[3];
  smallest_eigvec3(xx, xy, xz, yy, yz, zz, v);
  nrm[(size_t)g * 3 + 0] = v[0];
  nrm[(size_t)g * 3 + 1] = v[1];
  nrm[(size_t)g * 3 + 2] = v[2];
}

// ---------------------------------------------------------------------------
// Furthest point sampling: one workgroup per batch (serial outer loop),
// LDS argmax tree reduction each step.
// ---------------------------------------------------------------------------
__global__ void fps_kernel(const float* __restrict__ xyz, int* __restrict__ out,
                           float* __restrict__ dists, int N, int npoint) {
  int b = blockIdx.x;
  const float* xb = xyz + (size_t)b * N * 3;
  int* ob = out + (size_t)b * npoint;
  float* db = dists + (size_t)b * N;
  __shared__ float smax[256];
  __shared__ int simax[256];
  __shared__ int s_far;
  int tid = threadIdx.x;
  for (int i = tid; i < N; i += 256) db[i] = 1e10f;
  if (tid == 0) { s_far = 0; ob[0] = 0; }
  __syncthreads();
  for (int it = 1; it < npoint; ++it) {
    int f = s_far;
    float cx = xb[(size_t)f * 3], cy = xb[(size_t)f * 3 + 1], cz = xb[(size_t)f * 3 + 2];
    float bm = -1.f; int bmi = 0;
    for (int i = tid; i < N; i += 256) {
      float dx = xb[(size_t)i * 3] - cx;
      float dy = xb[(size_t)i * 3 + 1] - cy;
      float dz = xb[(size_t)i * 3 + 2] - cz;
      float d = dx * dx + dy * dy + dz * dz;
      float nd = fminf(db[i], d);
      db[i] = nd;
      if (nd > bm) { bm = nd; bmi = i; }
    }
    smax[tid] = bm; simax[tid] = bmi;
    __syncthreads();
    for (int s = 128; s > 0; s >>= 1) {
      if (tid < s) {
        bool take = (smax[tid + s] > smax[tid]) ||
                    (smax[tid + s] == smax[tid] && simax[tid + s] < simax[tid]);
        if (take) { smax[tid] = smax[tid + s]; simax[tid] = simax[tid + s]; }
      }
      __syncthreads();
    }
    if (tid == 0) { s_far = simax[0]; ob[it] = simax[0]; }
    __syncthreads();
  }
}

// ---------------------------------------------------------------------------
// generic gather / concat / elementwise kernels
// ---------------------------------------------------------------------------
__global__ void gather_rows_kernel(const float* __restrict__ src, const int* __restrict__ idx,
                                   float* __restrict__ dst, int Nsrc, int Ndst, int C) {
  int g = blockIdx.x * blockDim.x + threadIdx.x;
  int total = BATCH * Ndst * C;
  if (g >= total) return;
  int c = g % C; int t = g / C; int n = t % Ndst; int b = t / Ndst;
  int s = idx[(size_t)b * Ndst + n];
  dst[g] = src[((size_t)b * Nsrc + s) * C + c];
}

__global__ void concat2_kernel(const float* __restrict__ a, const float* __restrict__ b,
                               float* __restrict__ out, int rows, int Ca, int Cb) {
  int Ct = Ca + Cb;
  int g = blockIdx.x * blockDim.x + threadIdx.x;
  int total = rows * Ct;
  if (g >= total) return;
  int c = g % Ct; int r = g / Ct;
  out[g] = (c < Ca) ? a[(size_t)r * Ca + c] : b[(size_t)r * Cb + (c - Ca)];
}

__global__ void add_kernel(const float* __restrict__ a, const float* __restrict__ b,
                           float* __restrict__ out, int n) {
  int g = blockIdx.x * blockDim.x + threadIdx.x;
  if (g < n) out[g] = a[g] + b[g];
}

// feat = [gathered_xyz - new_xyz , gathered_points]  -> (B*np*K, 3+Cpts)
__global__ void sa_feat_kernel(const float* __restrict__ xyz, const float* __restrict__ nxyz,
                               const float* __restrict__ pts, const int* __restrict__ nidx,
                               float* __restrict__ out, int Nsrc, int np, int Cpts) {
  int Cout = 3 + Cpts;
  int g = blockIdx.x * blockDim.x + threadIdx.x;
  int total = BATCH * np * KNB * Cout;
  if (g >= total) return;
  int c = g % Cout; int t = g / Cout; int j = t % KNB;
  int t2 = t / KNB; int n = t2 % np; int b = t2 / np;
  int s = nidx[((size_t)b * np + n) * KNB + j];
  float v;
  if (c < 3)
    v = xyz[((size_t)b * Nsrc + s) * 3 + c] - nxyz[((size_t)b * np + n) * 3 + c];
  else
    v = pts[((size_t)b * Nsrc + s) * Cpts + (c - 3)];
  out[g] = v;
}

__global__ void maxpool_k_kernel(const float* __restrict__ in, float* __restrict__ out,
                                 int rows, int C) {
  int g = blockIdx.x * blockDim.x + threadIdx.x;
  int total = rows * C;
  if (g >= total) return;
  int c = g % C; int r = g / C;
  const float* p = in + ((size_t)r * KNB) * C + c;
  float m = p[0];
#pragma unroll
  for (int j = 1; j < KNB; ++j) m = fmaxf(m, p[(size_t)j * C]);
  out[g] = m;
}

__global__ void pos_rel_kernel(const float* __restrict__ pos, const int* __restrict__ idx,
                               float* __restrict__ out, int N) {
  int g = blockIdx.x * blockDim.x + threadIdx.x;
  int total = BATCH * N * KNB * 6;
  if (g >= total) return;
  int c = g % 6; int t = g / 6; int j = t % KNB; int bn = t / KNB; int b = bn / N;
  int s = idx[(size_t)bn * KNB + j];
  out[g] = pos[(size_t)bn * 6 + c] - pos[((size_t)b * N + s) * 6 + c];
}

// t = q[bn] - k_gathered + pos_emb
__global__ void qk_pos_kernel(const float* __restrict__ q, const float* __restrict__ kg,
                              const float* __restrict__ pe, float* __restrict__ out,
                              int N, int C) {
  int g = blockIdx.x * blockDim.x + threadIdx.x;
  int total = BATCH * N * KNB * C;
  if (g >= total) return;
  int c = g % C; int t = g / C; int bn = t / KNB;
  out[g] = q[(size_t)bn * C + c] - kg[g] + pe[g];
}

// softmax over neighbors + aggregation of (v[idx] + pos_emb)
__global__ void attn_agg_kernel(const float* __restrict__ attn, const float* __restrict__ v,
                                const float* __restrict__ pe, const int* __restrict__ idx,
                                float* __restrict__ out, int N, int C) {
  int g = blockIdx.x * blockDim.x + threadIdx.x;
  int total = BATCH * N * C;
  if (g >= total) return;
  int c = g % C; int bn = g / C; int b = bn / N;
  const float* ap = attn + (size_t)bn * KNB * C + c;
  const float* pep = pe + (size_t)bn * KNB * C + c;
  const int* ip = idx + (size_t)bn * KNB;
  float mx = -1e30f;
#pragma unroll
  for (int j = 0; j < KNB; ++j) mx = fmaxf(mx, ap[(size_t)j * C]);
  float ssum = 0.f, acc = 0.f;
#pragma unroll
  for (int j = 0; j < KNB; ++j) {
    float e = __expf(ap[(size_t)j * C] - mx);
    float vv = v[((size_t)b * N + ip[j]) * C + c] + pep[(size_t)j * C];
    ssum += e;
    acc += e * vv;
  }
  out[g] = acc / ssum;
}

// 3-NN inverse-distance interpolation, concat with skip features
__global__ void fp_interp_kernel(const float* __restrict__ pts2, const int* __restrict__ idx3,
                                 const float* __restrict__ dist3, const float* __restrict__ pts1,
                                 float* __restrict__ out, int N1v, int N2v, int C2, int C1) {
  int Ct = C2 + C1;
  int g = blockIdx.x * blockDim.x + threadIdx.x;
  int total = BATCH * N1v * Ct;
  if (g >= total) return;
  int c = g % Ct; int bn = g / Ct; int b = bn / N1v;
  if (c < C2) {
    const int* ip = idx3 + (size_t)bn * 3;
    const float* dp = dist3 + (size_t)bn * 3;
    float r0 = 1.f / fmaxf(dp[0], 1e-10f);
    float r1 = 1.f / fmaxf(dp[1], 1e-10f);
    float r2 = 1.f / fmaxf(dp[2], 1e-10f);
    float ws = r0 + r1 + r2;
    float v = r0 * pts2[((size_t)b * N2v + ip[0]) * C2 + c] +
              r1 * pts2[((size_t)b * N2v + ip[1]) * C2 + c] +
              r2 * pts2[((size_t)b * N2v + ip[2]) * C2 + c];
    out[g] = v / ws;
  } else {
    out[g] = pts1[(size_t)bn * C1 + (c - C2)];
  }
}

// ---------------------------------------------------------------------------
// weight prep: W (Cin x Cout, f32) -> Wt (Cout x CinP, f16, zero-padded K)
// ---------------------------------------------------------------------------
__global__ void wt_conv_kernel(const float* __restrict__ W, _Float16* __restrict__ Wt,
                               int Cin, int CinP, int Cout) {
  int g = blockIdx.x * blockDim.x + threadIdx.x;
  int total = Cout * CinP;
  if (g >= total) return;
  int k = g % CinP, n = g / CinP;
  float v = (k < Cin) ? W[(size_t)k * Cout + n] : 0.f;
  Wt[g] = (_Float16)v;
}

// ---------------------------------------------------------------------------
// WMMA dense layer: Y = act(X @ W + bias), f16 math / f32 accumulate.
// One wave owns a 16x64 output strip (4 accumulators, A fragment reused 4x).
// Branch-free straight-line inner loop: all 4 wmmas always issue (clamped
// pointers make OOB loads safe; OOB results never stored) so accumulators
// stay in-place (D==C). A fast/slow path hoisted out of the K loop; in the
// generic path the K-tail loads are unconditional from clamped (always
// in-bounds) addresses followed by a select -> no exec-mask branches.
// Latency hidden by occupancy (no LDS, no barriers).
// Fragment layout per CDNA5 ISA 7.12.2:
//   A: lane l<16 -> row l, K {0..7,16..23}; l>=16 -> K {8..15,24..31}
//   B: lane l -> col l%16, same K split;  C/D: col l%16, rows (l/16)*8+i
// ---------------------------------------------------------------------------
__device__ __forceinline__ v16h load_b_frag(const _Float16* __restrict__ Wrow,
                                            int ka, int kb) {
  v8h b0 = *(const v8h*)(Wrow + ka);
  v8h b1 = *(const v8h*)(Wrow + kb);
  v16h bf;
#pragma unroll
  for (int i = 0; i < 8; ++i) { bf[i] = b0[i]; bf[8 + i] = b1[i]; }
  return bf;
}

__global__ void dense_wmma_kernel(const float* __restrict__ X, const _Float16* __restrict__ Wt,
                                  const float* __restrict__ bias, float* __restrict__ Y,
                                  int R, int Cin, int CinP, int Cout, int relu) {
  const int lane = threadIdx.x & 31;
  const int wv = blockIdx.x * ((int)blockDim.x >> 5) + ((int)threadIdx.x >> 5);
  const int tilesM = (R + 15) >> 4;
  const int groupsN = (Cout + 63) >> 6;
  if (wv >= tilesM * groupsN) return;
  const int tm = wv % tilesM;
  const int tg = wv / tilesM;
  const int half = lane >> 4;
  const int mn = lane & 15;
  const int row = tm * 16 + mn;
  const int rowc = (row < R) ? row : (R - 1);
  const float* Xrow = X + (size_t)rowc * Cin;
  const int nbase = tg * 64;
  const _Float16* Wrow[4];
  int ncol[4];
#pragma unroll
  for (int u = 0; u < 4; ++u) {
    int n = nbase + u * 16 + mn;
    ncol[u] = n;
    int nc = (n < Cout) ? n : (Cout - 1);
    Wrow[u] = Wt + (size_t)nc * CinP;
  }
  v8f zero = {};
  v8f acc[4] = {zero, zero, zero, zero};
  if ((Cin & 31) == 0) {
    // fast path: every K step is full & 16B-aligned -> pure b128 traffic
    for (int k0 = 0; k0 < CinP; k0 += 32) {
      const int ka = k0 + half * 8;
      const int kb = k0 + 16 + half * 8;
      const v4f* p0 = (const v4f*)(Xrow + ka);
      const v4f* p1 = (const v4f*)(Xrow + kb);
      v4f a0 = p0[0], a1 = p0[1], a2 = p1[0], a3 = p1[1];
      v16h af;
#pragma unroll
      for (int i = 0; i < 4; ++i) {
        af[i]      = (_Float16)a0[i];
        af[4 + i]  = (_Float16)a1[i];
        af[8 + i]  = (_Float16)a2[i];
        af[12 + i] = (_Float16)a3[i];
      }
#pragma unroll
      for (int u = 0; u < 4; ++u) {
        v16h bf = load_b_frag(Wrow[u], ka, kb);
        acc[u] = __builtin_amdgcn_wmma_f32_16x16x32_f16(
            false, af, false, bf, (short)0, acc[u], false, false);
      }
    }
  } else {
    // generic path: unconditional clamped loads + selects (no branches)
    for (int k0 = 0; k0 < CinP; k0 += 32) {
      const int ka = k0 + half * 8;
      const int kb = k0 + 16 + half * 8;
      v16h af;
#pragma unroll
      for (int e = 0; e < 8; ++e) {
        int k1 = ka + e, k2 = kb + e;
        int k1c = (k1 < Cin) ? k1 : 0;   // clamped -> always in-bounds
        int k2c = (k2 < Cin) ? k2 : 0;
        float va = Xrow[k1c];            // unconditional load
        float vb = Xrow[k2c];
        va = (k1 < Cin) ? va : 0.f;      // branchless zero for K pad
        vb = (k2 < Cin) ? vb : 0.f;
        af[e]     = (_Float16)va;
        af[8 + e] = (_Float16)vb;
      }
#pragma unroll
      for (int u = 0; u < 4; ++u) {
        v16h bf = load_b_frag(Wrow[u], ka, kb);
        acc[u] = __builtin_amdgcn_wmma_f32_16x16x32_f16(
            false, af, false, bf, (short)0, acc[u], false, false);
      }
    }
  }
#pragma unroll
  for (int u = 0; u < 4; ++u) {
    int n = ncol[u];
    if (n < Cout) {
      float bv = bias[n];
#pragma unroll
      for (int i = 0; i < 8; ++i) {
        int m = tm * 16 + half * 8 + i;
        if (m < R) {
          float v = acc[u][i] + bv;
          if (relu) v = fmaxf(v, 0.f);
          Y[(size_t)m * Cout + n] = v;
        }
      }
    }
  }
}

// ---------------------------------------------------------------------------
// final: sigmoid score, mask (<0.7), write all three transposed outputs
// out layout: [score (B,1,N)] [cleaned (B,3,N)] [pw (B,128,N)]
// ---------------------------------------------------------------------------
__global__ void final_kernel(const float* __restrict__ logit, const float* __restrict__ xyz,
                             const float* __restrict__ feat, float* __restrict__ out) {
  int g = blockIdx.x * blockDim.x + threadIdx.x;
  if (g >= BATCH * NPTS) return;
  int b = g / NPTS, n = g % NPTS;
  float s = 1.f / (1.f + __expf(-logit[g]));
  float mask = (s < 0.7f) ? 1.f : 0.f;
  out[(size_t)b * NPTS + n] = s;
  float* oc = out + (size_t)BATCH * NPTS;
#pragma unroll
  for (int c = 0; c < 3; ++c)
    oc[((size_t)b * 3 + c) * NPTS + n] = xyz[((size_t)b * NPTS + n) * 3 + c] * mask;
  float* op = oc + (size_t)BATCH * 3 * NPTS;
  for (int c = 0; c < 128; ++c)
    op[((size_t)b * 128 + c) * NPTS + n] = feat[((size_t)b * NPTS + n) * 128 + c] * mask;
}

// ---------------------------------------------------------------------------
// host orchestration
// ---------------------------------------------------------------------------
extern "C" void kernel_launch(void* const* d_in, const int* in_sizes, int n_in,
                              void* d_out, int out_size, void* d_ws, size_t ws_size,
                              hipStream_t stream) {
  (void)in_sizes; (void)out_size; (void)ws_size;
  const float* P[64];
  for (int i = 0; i < 64; ++i) P[i] = (i < n_in) ? (const float*)d_in[i] : nullptr;

  // setup_inputs() dict insertion order: xyz, then params tree, each lin = (W, b)
  const float* xyz = P[0];
  const float *sa1w0 = P[1], *sa1b0 = P[2], *sa1w1 = P[3], *sa1b1 = P[4];
  const float *t1ws = P[5], *t1bs = P[6], *t1wq = P[7], *t1bq = P[8],
              *t1wk = P[9], *t1bk = P[10], *t1wv = P[11], *t1bv = P[12],
              *t1wp0 = P[13], *t1bp0 = P[14], *t1wp1 = P[15], *t1bp1 = P[16],
              *t1wa0 = P[17], *t1ba0 = P[18], *t1wa1 = P[19], *t1ba1 = P[20],
              *t1we = P[21], *t1be = P[22];
  const float *sa2w0 = P[23], *sa2b0 = P[24], *sa2w1 = P[25], *sa2b1 = P[26];
  const float *t2ws = P[27], *t2bs = P[28], *t2wq = P[29], *t2bq = P[30],
              *t2wk = P[31], *t2bk = P[32], *t2wv = P[33], *t2bv = P[34],
              *t2wp0 = P[35], *t2bp0 = P[36], *t2wp1 = P[37], *t2bp1 = P[38],
              *t2wa0 = P[39], *t2ba0 = P[40], *t2wa1 = P[41], *t2ba1 = P[42],
              *t2we = P[43], *t2be = P[44];
  const float *fp1w0 = P[45], *fp1b0 = P[46], *fp1w1 = P[47], *fp1b1 = P[48];
  const float *fp2w0 = P[49], *fp2b0 = P[50], *fp2w1 = P[51], *fp2b1 = P[52];
  const float *ow0 = P[53], *ob0 = P[54], *ow1 = P[55], *ob1 = P[56],
              *ow2 = P[57], *ob2 = P[58];

  // workspace carve-out
  char* wsbase = (char*)d_ws;
  size_t wsoff = 0;
  auto allocB = [&](size_t bytes) -> void* {
    void* p = wsbase + wsoff;
    wsoff += (bytes + 255) & ~(size_t)255;
    return p;
  };
  auto allocF = [&](size_t n) -> float* { return (float*)allocB(n * sizeof(float)); };
  auto allocI = [&](size_t n) -> int* { return (int*)allocB(n * sizeof(int)); };

  int* nidx0 = allocI((size_t)BATCH * NPTS * NORMK);
  float* nrm = allocF((size_t)BATCH * NPTS * 3);
  float* l0p = allocF((size_t)BATCH * NPTS * 6);
  float* fdist = allocF((size_t)BATCH * NPTS);
  int* fps1 = allocI((size_t)BATCH * NP1);
  float* l1x = allocF((size_t)BATCH * NP1 * 3);
  float* l1n = allocF((size_t)BATCH * NP1 * 3);
  int* sa1n = allocI((size_t)BATCH * NP1 * KNB);
  float* sa1f = allocF((size_t)BATCH * NP1 * KNB * 9);
  float* sa1h = allocF((size_t)BATCH * NP1 * KNB * 64);
  float* sa1o = allocF((size_t)BATCH * NP1 * KNB * 128);
  float* l1p = allocF((size_t)BATCH * NP1 * 128);
  // transformer 1 scratch
  float* pos1 = allocF((size_t)BATCH * NP1 * 6);
  int* t1i = allocI((size_t)BATCH * NP1 * KNB);
  float* t1x = allocF((size_t)BATCH * NP1 * 64);
  float* t1q = allocF((size_t)BATCH * NP1 * 64);
  float* t1k = allocF((size_t)BATCH * NP1 * 64);
  float* t1v = allocF((size_t)BATCH * NP1 * 64);
  float* t1kg = allocF((size_t)BATCH * NP1 * KNB * 64);
  float* t1pr = allocF((size_t)BATCH * NP1 * KNB * 6);
  float* t1ph = allocF((size_t)BATCH * NP1 * KNB * 64);
  float* t1pe = allocF((size_t)BATCH * NP1 * KNB * 64);
  float* t1ts = allocF((size_t)BATCH * NP1 * KNB * 64);
  float* t1ah = allocF((size_t)BATCH * NP1 * KNB * 256);
  float* t1at = allocF((size_t)BATCH * NP1 * KNB * 64);
  float* t1ag = allocF((size_t)BATCH * NP1 * 64);
  float* t1o = allocF((size_t)BATCH * NP1 * 128);
  float* l1p2 = allocF((size_t)BATCH * NP1 * 128);
  // SA2
  int* fps2 = allocI((size_t)BATCH * NP2);
  float* l2x = allocF((size_t)BATCH * NP2 * 3);
  float* l2n = allocF((size_t)BATCH * NP2 * 3);
  int* sa2n = allocI((size_t)BATCH * NP2 * KNB);
  float* sa2f = allocF((size_t)BATCH * NP2 * KNB * 131);
  float* sa2h = allocF((size_t)BATCH * NP2 * KNB * 128);
  float* sa2o = allocF((size_t)BATCH * NP2 * KNB * 256);
  float* l2p = allocF((size_t)BATCH * NP2 * 256);
  // transformer 2 scratch
  float* pos2 = allocF((size_t)BATCH * NP2 * 6);
  int* t2i = allocI((size_t)BATCH * NP2 * KNB);
  float* t2x = allocF((size_t)BATCH * NP2 * 64);
  float* t2q = allocF((size_t)BATCH * NP2 * 64);
  float* t2k = allocF((size_t)BATCH * NP2 * 64);
  float* t2v = allocF((size_t)BATCH * NP2 * 64);
  float* t2kg = allocF((size_t)BATCH * NP2 * KNB * 64);
  float* t2pr = allocF((size_t)BATCH * NP2 * KNB * 6);
  float* t2ph = allocF((size_t)BATCH * NP2 * KNB * 64);
  float* t2pe = allocF((size_t)BATCH * NP2 * KNB * 64);
  float* t2ts = allocF((size_t)BATCH * NP2 * KNB * 64);
  float* t2ah = allocF((size_t)BATCH * NP2 * KNB * 256);
  float* t2at = allocF((size_t)BATCH * NP2 * KNB * 64);
  float* t2ag = allocF((size_t)BATCH * NP2 * 64);
  float* t2o = allocF((size_t)BATCH * NP2 * 256);
  float* l2p2 = allocF((size_t)BATCH * NP2 * 256);
  // FP modules
  int* fp1i = allocI((size_t)BATCH * NP1 * 3);
  float* fp1d = allocF((size_t)BATCH * NP1 * 3);
  float* fp1c = allocF((size_t)BATCH * NP1 * 384);
  float* fp1h = allocF((size_t)BATCH * NP1 * 256);
  float* l1f = allocF((size_t)BATCH * NP1 * 256);
  int* fp2i = allocI((size_t)BATCH * NPTS * 3);
  float* fp2d = allocF((size_t)BATCH * NPTS * 3);
  float* fp2c = allocF((size_t)BATCH * NPTS * 262);
  float* fp2h = allocF((size_t)BATCH * NPTS * 128);
  float* l0f = allocF((size_t)BATCH * NPTS * 128);
  // output MLP
  float* o1 = allocF((size_t)BATCH * NPTS * 64);
  float* o2 = allocF((size_t)BATCH * NPTS * 8);
  float* logit = allocF((size_t)BATCH * NPTS);

  // dense layer: f16 weight prep + barrier-free WMMA GEMM
  auto dense_l = [&](const float* Xp, const float* Wp, const float* bp, float* Yp,
                     int R, int Cin, int Cout, int relu) {
    int CinP = (Cin + 31) & ~31;
    _Float16* Wt = (_Float16*)allocB((size_t)Cout * CinP * sizeof(_Float16));
    wt_conv_kernel<<<cdiv(Cout * CinP, 256), 256, 0, stream>>>(Wp, Wt, Cin, CinP, Cout);
    int tilesM = cdiv(R, 16);
    int groupsN = cdiv(Cout, 64);
    int waves = tilesM * groupsN;
    dense_wmma_kernel<<<cdiv(waves, 8), 256, 0, stream>>>(Xp, Wt, bp, Yp, R, Cin, CinP,
                                                          Cout, relu);
  };

  auto run_transformer = [&](const float* ptsin, float* ptsout, const float* xyzp,
                             const float* nrmp, int N, int Cp,
                             const float* Ws, const float* bs, const float* Wq, const float* bq,
                             const float* Wk, const float* bk, const float* Wv, const float* bv,
                             const float* Wp0, const float* bp0, const float* Wp1, const float* bp1,
                             const float* Wa0, const float* ba0, const float* Wa1, const float* ba1,
                             const float* We, const float* be, float* posb, int* tidx,
                             float* xb, float* qb, float* kb, float* vb, float* kg, float* pr,
                             float* ph, float* pe, float* ts, float* ah, float* at, float* ag,
                             float* to_) {
    int R = BATCH * N;
    int Rk = R * KNB;
    concat2_kernel<<<cdiv(R * 6, 256), 256, 0, stream>>>(xyzp, nrmp, posb, R, 3, 3);
    knn_kernel<KNB><<<cdiv(R, 128), 128, 0, stream>>>(xyzp, xyzp, N, N, 3, tidx, nullptr);
    dense_l(ptsin, Ws, bs, xb, R, Cp, 64, 0);
    dense_l(xb, Wq, bq, qb, R, 64, 64, 0);
    dense_l(xb, Wk, bk, kb, R, 64, 64, 0);
    dense_l(xb, Wv, bv, vb, R, 64, 64, 0);
    gather_rows_kernel<<<cdiv(Rk * 64, 256), 256, 0, stream>>>(kb, tidx, kg, N, N * KNB, 64);
    pos_rel_kernel<<<cdiv(Rk * 6, 256), 256, 0, stream>>>(posb, tidx, pr, N);
    dense_l(pr, Wp0, bp0, ph, Rk, 6, 64, 1);
    dense_l(ph, Wp1, bp1, pe, Rk, 64, 64, 0);
    qk_pos_kernel<<<cdiv(Rk * 64, 256), 256, 0, stream>>>(qb, kg, pe, ts, N, 64);
    dense_l(ts, Wa0, ba0, ah, Rk, 64, 256, 1);
    dense_l(ah, Wa1, ba1, at, Rk, 256, 64, 0);
    attn_agg_kernel<<<cdiv(R * 64, 256), 256, 0, stream>>>(at, vb, pe, tidx, ag, N, 64);
    dense_l(ag, We, be, to_, R, 64, Cp, 0);
    add_kernel<<<cdiv(R * Cp, 256), 256, 0, stream>>>(to_, ptsin, ptsout, R * Cp);
  };

  // ---- stage 0: normals + l0 features -------------------------------------
  knn_kernel<NORMK><<<cdiv(BATCH * NPTS, 128), 128, 0, stream>>>(
      xyz, xyz, NPTS, NPTS, 3, nidx0, nullptr);
  cal_axis_kernel<<<cdiv(BATCH * NPTS, 256), 256, 0, stream>>>(xyz, nidx0, nrm);
  concat2_kernel<<<cdiv(BATCH * NPTS * 6, 256), 256, 0, stream>>>(
      xyz, nrm, l0p, BATCH * NPTS, 3, 3);

  // ---- SA1 ----------------------------------------------------------------
  fps_kernel<<<BATCH, 256, 0, stream>>>(xyz, fps1, fdist, NPTS, NP1);
  gather_rows_kernel<<<cdiv(BATCH * NP1 * 3, 256), 256, 0, stream>>>(xyz, fps1, l1x, NPTS, NP1, 3);
  gather_rows_kernel<<<cdiv(BATCH * NP1 * 3, 256), 256, 0, stream>>>(nrm, fps1, l1n, NPTS, NP1, 3);
  knn_kernel<KNB><<<cdiv(BATCH * NP1, 128), 128, 0, stream>>>(l1x, xyz, NP1, NPTS, 3, sa1n, nullptr);
  sa_feat_kernel<<<cdiv(BATCH * NP1 * KNB * 9, 256), 256, 0, stream>>>(
      xyz, l1x, l0p, sa1n, sa1f, NPTS, NP1, 6);
  dense_l(sa1f, sa1w0, sa1b0, sa1h, BATCH * NP1 * KNB, 9, 64, 1);
  dense_l(sa1h, sa1w1, sa1b1, sa1o, BATCH * NP1 * KNB, 64, 128, 0);
  maxpool_k_kernel<<<cdiv(BATCH * NP1 * 128, 256), 256, 0, stream>>>(sa1o, l1p, BATCH * NP1, 128);

  // ---- transformer 1 ------------------------------------------------------
  run_transformer(l1p, l1p2, l1x, l1n, NP1, 128,
                  t1ws, t1bs, t1wq, t1bq, t1wk, t1bk, t1wv, t1bv,
                  t1wp0, t1bp0, t1wp1, t1bp1, t1wa0, t1ba0, t1wa1, t1ba1, t1we, t1be,
                  pos1, t1i, t1x, t1q, t1k, t1v, t1kg, t1pr, t1ph, t1pe, t1ts,
                  t1ah, t1at, t1ag, t1o);

  // ---- SA2 ----------------------------------------------------------------
  fps_kernel<<<BATCH, 256, 0, stream>>>(l1x, fps2, fdist, NP1, NP2);
  gather_rows_kernel<<<cdiv(BATCH * NP2 * 3, 256), 256, 0, stream>>>(l1x, fps2, l2x, NP1, NP2, 3);
  gather_rows_kernel<<<cdiv(BATCH * NP2 * 3, 256), 256, 0, stream>>>(l1n, fps2, l2n, NP1, NP2, 3);
  knn_kernel<KNB><<<cdiv(BATCH * NP2, 128), 128, 0, stream>>>(l2x, l1x, NP2, NP1, 3, sa2n, nullptr);
  sa_feat_kernel<<<cdiv(BATCH * NP2 * KNB * 131, 256), 256, 0, stream>>>(
      l1x, l2x, l1p2, sa2n, sa2f, NP1, NP2, 128);
  dense_l(sa2f, sa2w0, sa2b0, sa2h, BATCH * NP2 * KNB, 131, 128, 1);
  dense_l(sa2h, sa2w1, sa2b1, sa2o, BATCH * NP2 * KNB, 128, 256, 0);
  maxpool_k_kernel<<<cdiv(BATCH * NP2 * 256, 256), 256, 0, stream>>>(sa2o, l2p, BATCH * NP2, 256);

  // ---- transformer 2 ------------------------------------------------------
  run_transformer(l2p, l2p2, l2x, l2n, NP2, 256,
                  t2ws, t2bs, t2wq, t2bq, t2wk, t2bk, t2wv, t2bv,
                  t2wp0, t2bp0, t2wp1, t2bp1, t2wa0, t2ba0, t2wa1, t2ba1, t2we, t2be,
                  pos2, t2i, t2x, t2q, t2k, t2v, t2kg, t2pr, t2ph, t2pe, t2ts,
                  t2ah, t2at, t2ag, t2o);

  // ---- FP1: l2 -> l1 ------------------------------------------------------
  knn_kernel<3><<<cdiv(BATCH * NP1, 128), 128, 0, stream>>>(pos1, pos2, NP1, NP2, 6, fp1i, fp1d);
  fp_interp_kernel<<<cdiv(BATCH * NP1 * 384, 256), 256, 0, stream>>>(
      l2p2, fp1i, fp1d, l1p2, fp1c, NP1, NP2, 256, 128);
  dense_l(fp1c, fp1w0, fp1b0, fp1h, BATCH * NP1, 384, 256, 1);
  dense_l(fp1h, fp1w1, fp1b1, l1f, BATCH * NP1, 256, 256, 1);

  // ---- FP2: l1 -> l0 ------------------------------------------------------
  knn_kernel<3><<<cdiv(BATCH * NPTS, 128), 128, 0, stream>>>(l0p, pos1, NPTS, NP1, 6, fp2i, fp2d);
  fp_interp_kernel<<<cdiv(BATCH * NPTS * 262, 256), 256, 0, stream>>>(
      l1f, fp2i, fp2d, l0p, fp2c, NPTS, NP1, 256, 6);
  dense_l(fp2c, fp2w0, fp2b0, fp2h, BATCH * NPTS, 262, 128, 1);
  dense_l(fp2h, fp2w1, fp2b1, l0f, BATCH * NPTS, 128, 128, 1);

  // ---- output head --------------------------------------------------------
  dense_l(l0f, ow0, ob0, o1, BATCH * NPTS, 128, 64, 1);
  dense_l(o1, ow1, ob1, o2, BATCH * NPTS, 64, 8, 1);
  dense_l(o2, ow2, ob2, logit, BATCH * NPTS, 8, 1, 0);
  final_kernel<<<cdiv(BATCH * NPTS, 256), 256, 0, stream>>>(logit, xyz, l0f, (float*)d_out);
}
